// GRUModule_49435073577342
// MI455X (gfx1250) — matrix operand order repaired and checked by reading, not compiled, not run on hardware
//
#include <hip/hip_runtime.h>
#include <hip/hip_bf16.h>

// GRU scan for MI455X (gfx1250): persistent kernel, bf16 WMMA, bf16 h ping-pong
// (A-fragments = direct global_load_b128, no per-step cvt), LDS-resident
// pre-swizzled weight B-fragments, async-to-LDS double-buffered x_t prefetch
// (ASYNCcnt), grid-wide sense barrier per timestep.

typedef __attribute__((ext_vector_type(16))) __bf16 v16bf;
typedef __attribute__((ext_vector_type(8)))  __bf16 v8bf;
typedef __attribute__((ext_vector_type(4)))  __bf16 v4bf;
typedef __attribute__((ext_vector_type(8)))  float  v8f;

#define GRU_N 64
#define GRU_T 2048
#define GRU_D 128
#define GRU_H 256
#define NWG   16          // workgroups = H / JW
#define JW    16          // hidden units per workgroup
#define NTHREADS 128      // 4 waves (wave32)

// dynamic LDS layout (bytes)
#define SM_BHH 0          // 3*8*32 v16bf          = 24576
#define SM_BIH 24576      // 3*4*32 v16bf          = 12288
#define SM_XA  36864      // 2 x 64x128 bf16       = 32768
#define SM_SR  69632      // 64x16 f32             =  4096
#define SM_SZ  73728
#define SM_SHN 77824
#define SM_SIN 81920
#define SM_HSL 86016      // 64x16 f32 own h slice =  4096
#define SM_TOTAL 90112

__device__ __forceinline__ float sigmoidf_(float v) {
    return 1.0f / (1.0f + __expf(-v));
}

__device__ __forceinline__ v8f splat8(float s) {
    return (v8f){s, s, s, s, s, s, s, s};
}

__device__ __forceinline__ v16bf cat8(v8bf lo, v8bf hi) {
    return __builtin_shufflevector(lo, hi, 0, 1, 2, 3, 4, 5, 6, 7,
                                           8, 9, 10, 11, 12, 13, 14, 15);
}

__device__ __forceinline__ void grid_barrier(unsigned* cnt, unsigned* sense, unsigned expect) {
    __syncthreads();
    if (threadIdx.x == 0) {
        __threadfence();
        unsigned arrived =
            __hip_atomic_fetch_add(cnt, 1u, __ATOMIC_ACQ_REL, __HIP_MEMORY_SCOPE_AGENT);
        if (arrived == (unsigned)(NWG - 1)) {
            __hip_atomic_store(cnt, 0u, __ATOMIC_RELAXED, __HIP_MEMORY_SCOPE_AGENT);
            __hip_atomic_store(sense, expect, __ATOMIC_RELEASE, __HIP_MEMORY_SCOPE_AGENT);
        } else {
            while (__hip_atomic_load(sense, __ATOMIC_ACQUIRE, __HIP_MEMORY_SCOPE_AGENT) < expect) {
                __builtin_amdgcn_s_sleep(2);
            }
        }
        __threadfence();
    }
    __syncthreads();
}

// per-lane async copy: 16 bytes global -> LDS, tracked by ASYNCcnt
__device__ __forceinline__ void async_b128_to_lds(unsigned lds_byte_off, const void* gaddr) {
    asm volatile("global_load_async_to_lds_b128 %0, %1, off"
                 :: "v"(lds_byte_off),
                    "v"((unsigned long long)(uintptr_t)gaddr)
                 : "memory");
}

// issue one 64x128 bf16 x-tile (16KB) into XA[buf]; 8 chunks of 16B per thread
__device__ __forceinline__ void prefetch_xtile(const __bf16* __restrict__ xb,
                                               __bf16* XA, int buf, int t1, int tid) {
    #pragma unroll
    for (int j = 0; j < 8; ++j) {
        int c  = tid + j * NTHREADS;      // chunk id, 1024 chunks
        int m  = c >> 4;                  // 16 chunks per row
        int kk = (c & 15) * 8;
        const __bf16* g = xb + (size_t)m * GRU_T * GRU_D + (size_t)t1 * GRU_D + kk;
        unsigned lds = (unsigned)(uintptr_t)(XA + (size_t)buf * (GRU_N * GRU_D) + c * 8);
        async_b128_to_lds(lds, g);
    }
}

// phase 0: convert x fp32 -> bf16 (fully parallel, HBM-bandwidth bound)
__global__ __launch_bounds__(256)
void cvt_x_kernel(const float* __restrict__ x, __bf16* __restrict__ xb, int n4) {
    int i = blockIdx.x * blockDim.x + threadIdx.x;
    if (i < n4) {
        const float4 f = ((const float4*)x)[i];
        v4bf o = {(__bf16)f.x, (__bf16)f.y, (__bf16)f.z, (__bf16)f.w};
        ((v4bf*)xb)[i] = o;
    }
}

__global__ __launch_bounds__(NTHREADS)
void gru_scan_kernel(const __bf16* __restrict__ xb,
                     const float* __restrict__ W_ih,
                     const float* __restrict__ W_hh,
                     const float* __restrict__ b_ih,
                     const float* __restrict__ b_hh,
                     float* __restrict__ out,
                     __bf16* __restrict__ hbuf,       // 2 * N * H bf16 ping-pong
                     unsigned* __restrict__ sync_cnt,
                     unsigned* __restrict__ sync_sense)
{
    extern __shared__ char smem[];
    v16bf*  Bhh = (v16bf*)(smem + SM_BHH);   // [gate(3)][kchunk(8)][lane(32)]
    v16bf*  Bih = (v16bf*)(smem + SM_BIH);   // [gate(3)][kchunk(4)][lane(32)]
    __bf16* XA  = (__bf16*)(smem + SM_XA);   // 2 x (64x128) x_t staging
    float*  S_r = (float*)(smem + SM_SR);
    float*  S_z = (float*)(smem + SM_SZ);
    float*  S_hn = (float*)(smem + SM_SHN);
    float*  S_in = (float*)(smem + SM_SIN);
    float*  Hsl = (float*)(smem + SM_HSL);   // own fp32 h slice (precision keeper)

    const int tid = threadIdx.x;
    const int j0  = blockIdx.x * JW;         // this WG's hidden-unit slice

    // ---- one-time: convert fp32 weight slices -> bf16 WMMA B-fragments in LDS
    {
        __bf16* bhh = (__bf16*)Bhh;
        for (int i = tid; i < 3 * 8 * 512; i += NTHREADS) {
            int tile = i >> 9, rem = i & 511;
            int lane = rem >> 4, q = rem & 15;
            int v = q >> 1, hh = q & 1;
            int nt = tile >> 3, kc = tile & 7;
            int col = lane & 15, khalf = lane >> 4;
            int row = nt * GRU_H + j0 + col;              // gate rows (r,z,n)
            int k   = kc * 32 + (v >> 2) * 16 + khalf * 8 + (v & 3) * 2 + hh;  // B layout: V v -> K=16(v/4)+8khalf+2(v%4)
            bhh[i] = (__bf16)W_hh[(size_t)row * GRU_H + k];
        }
        __bf16* bih = (__bf16*)Bih;
        for (int i = tid; i < 3 * 4 * 512; i += NTHREADS) {
            int tile = i >> 9, rem = i & 511;
            int lane = rem >> 4, q = rem & 15;
            int v = q >> 1, hh = q & 1;
            int nt = tile >> 2, kc = tile & 3;
            int col = lane & 15, khalf = lane >> 4;
            int row = nt * GRU_H + j0 + col;
            int k   = kc * 32 + (v >> 2) * 16 + khalf * 8 + (v & 3) * 2 + hh;
            bih[i] = (__bf16)W_ih[(size_t)row * GRU_D + k];
        }
        for (int i = tid; i < GRU_N * JW; i += NTHREADS) Hsl[i] = 0.0f;  // h0 = 0
    }

    // prefetch x_0 into XA[0] while we finish setup
    prefetch_xtile(xb, XA, 0, 0, tid);
    __syncthreads();

    const int lane  = tid & 31;
    const int wid   = tid >> 5;                // 4 waves -> 4 M-tiles (batch)
    const int mrow  = wid * 16 + (lane & 15);  // A-matrix row this lane feeds
    const int khalf = lane >> 4;               // A layout K-half select
    const int ccol  = lane & 15;               // C/D column (local hidden j)

    // biases folded into accumulator init (r,z get both; n keeps parts split)
    const float bR  = b_ih[j0 + ccol]             + b_hh[j0 + ccol];
    const float bZ  = b_ih[GRU_H + j0 + ccol]     + b_hh[GRU_H + j0 + ccol];
    const float bNH = b_hh[2 * GRU_H + j0 + ccol];
    const float bNI = b_ih[2 * GRU_H + j0 + ccol];

    for (int t = 0; t < GRU_T; ++t) {
        const __bf16* hsrc = hbuf + (size_t)(t & 1) * (GRU_N * GRU_H);
        __bf16*       hdst = hbuf + (size_t)((t + 1) & 1) * (GRU_N * GRU_H);

        // kick off next step's x tile (clamped at the end; ASYNCcnt += 8)
        int t1 = (t + 1 < GRU_T) ? (t + 1) : (GRU_T - 1);
        prefetch_xtile(xb, XA, (t + 1) & 1, t1, tid);

        v8f accR  = splat8(bR);
        v8f accZ  = splat8(bZ);
        v8f accNH = splat8(bNH);
        v8f accNI = splat8(bNI);

        // ---- recurrent GEMM slice: h(64x256) @ W_hh^T, K = 256
        // bf16 h in L2: A-fragment = two global b128 loads per k-chunk, no cvt
        const __bf16* ha = hsrc + (size_t)mrow * GRU_H;
        #pragma unroll
        for (int kc = 0; kc < 8; ++kc) {
            v8bf lo = *(const v8bf*)(ha + kc * 32 + khalf * 8);
            v8bf hi = *(const v8bf*)(ha + kc * 32 + 16 + khalf * 8);
            v16bf afr = cat8(lo, hi);
            v16bf b0 = Bhh[(0 * 8 + kc) * 32 + lane];
            v16bf b1 = Bhh[(1 * 8 + kc) * 32 + lane];
            v16bf b2 = Bhh[(2 * 8 + kc) * 32 + lane];
            accR  = __builtin_amdgcn_wmma_f32_16x16x32_bf16(false, afr, false, b0, (short)0, accR,  false, false);
            accZ  = __builtin_amdgcn_wmma_f32_16x16x32_bf16(false, afr, false, b1, (short)0, accZ,  false, false);
            accNH = __builtin_amdgcn_wmma_f32_16x16x32_bf16(false, afr, false, b2, (short)0, accNH, false, false);
        }

        // wait for x_t tile (8 newer asyncs may remain in flight; in-order done)
        asm volatile("s_wait_asynccnt 0x8" ::: "memory");
        __syncthreads();

        // ---- fused input GEMM slice: x_t(64x128) @ W_ih^T from LDS, K = 128
        const __bf16* xa = XA + (size_t)(t & 1) * (GRU_N * GRU_D) + (size_t)mrow * GRU_D;
        #pragma unroll
        for (int kc = 0; kc < 4; ++kc) {
            v8bf lo = *(const v8bf*)(xa + kc * 32 + khalf * 8);
            v8bf hi = *(const v8bf*)(xa + kc * 32 + 16 + khalf * 8);
            v16bf afr = cat8(lo, hi);
            v16bf b0 = Bih[(0 * 4 + kc) * 32 + lane];
            v16bf b1 = Bih[(1 * 4 + kc) * 32 + lane];
            v16bf b2 = Bih[(2 * 4 + kc) * 32 + lane];
            accR  = __builtin_amdgcn_wmma_f32_16x16x32_bf16(false, afr, false, b0, (short)0, accR,  false, false);
            accZ  = __builtin_amdgcn_wmma_f32_16x16x32_bf16(false, afr, false, b1, (short)0, accZ,  false, false);
            accNI = __builtin_amdgcn_wmma_f32_16x16x32_bf16(false, afr, false, b2, (short)0, accNI, false, false);
        }

        // ---- stage accumulators (C/D layout: VGPR v -> M = v + 8*khalf)
        #pragma unroll
        for (int v = 0; v < 8; ++v) {
            int m   = wid * 16 + khalf * 8 + v;
            int idx = m * JW + ccol;
            S_r [idx] = accR[v];
            S_z [idx] = accZ[v];
            S_hn[idx] = accNH[v];
            S_in[idx] = accNI[v];
        }
        __syncthreads();

        // ---- gate math + h update (h_prev slice kept fp32 in LDS)
        for (int i = tid; i < GRU_N * JW; i += NTHREADS) {
            int   m  = i >> 4, j = i & 15;
            float hp = Hsl[i];
            float r  = sigmoidf_(S_r[i]);
            float z  = sigmoidf_(S_z[i]);
            float nn = tanhf(S_in[i] + r * S_hn[i]);
            float hv = (1.0f - z) * nn + z * hp;
            Hsl[i] = hv;
            hdst[(size_t)m * GRU_H + j0 + j] = (__bf16)hv;   // bf16 for cross-WG GEMM
        }

        grid_barrier(sync_cnt, sync_sense, (unsigned)(t + 1));
    }

    // ---- final: out = tanh(h_T) for our slice, from the fp32 LDS copy
    for (int i = tid; i < GRU_N * JW; i += NTHREADS) {
        int m = i >> 4, j = i & 15;
        out[(size_t)m * GRU_H + j0 + j] = tanhf(Hsl[i]);
    }
}

extern "C" void kernel_launch(void* const* d_in, const int* in_sizes, int n_in,
                              void* d_out, int out_size, void* d_ws, size_t ws_size,
                              hipStream_t stream) {
    (void)in_sizes; (void)n_in; (void)out_size; (void)ws_size;

    const float* x    = (const float*)d_in[0];
    const float* W_ih = (const float*)d_in[1];
    const float* W_hh = (const float*)d_in[2];
    const float* b_ih = (const float*)d_in[3];
    const float* b_hh = (const float*)d_in[4];
    float*       out  = (float*)d_out;

    // workspace layout: [xb: N*T*D bf16][hbuf: 2*N*H bf16][sync cnt/sense]
    const size_t xbytes = (size_t)GRU_N * GRU_T * GRU_D * sizeof(__bf16); // 32 MB
    const size_t hbytes = (size_t)2 * GRU_N * GRU_H * sizeof(__bf16);     // 64 KB
    __bf16*   xb         = (__bf16*)d_ws;
    __bf16*   hbuf       = (__bf16*)((char*)d_ws + xbytes);
    unsigned* sync_cnt   = (unsigned*)((char*)d_ws + xbytes + hbytes);
    unsigned* sync_sense = sync_cnt + 1;

    // zero h0 + barrier state every launch (graph-capture safe, deterministic)
    hipMemsetAsync((char*)d_ws + xbytes, 0, hbytes + 64, stream);

    // phase 0: x -> bf16 (parallel, ~67MB read / 33MB write)
    const int n4 = (GRU_N * GRU_T * GRU_D) / 4;
    cvt_x_kernel<<<n4 / 256, 256, 0, stream>>>(x, xb, n4);

    // phase 1: persistent fused scan
    gru_scan_kernel<<<NWG, NTHREADS, SM_TOTAL, stream>>>(
        xb, W_ih, W_hh, b_ih, b_hh, out, hbuf, sync_cnt, sync_sense);
}
